// GCN_65240553226756
// MI455X (gfx1250) — matrix operand order repaired and checked by reading, not compiled, not run on hardware
//
#include <hip/hip_runtime.h>
#include <stdint.h>

typedef __attribute__((ext_vector_type(16))) _Float16 v16h;
typedef __attribute__((ext_vector_type(8)))  float    v8f;
typedef __attribute__((ext_vector_type(4)))  _Float16 v4h;

union Frag  { v16h v; uint4 q[2]; };
union Pack8 { _Float16 h[8]; uint4 q; };

__device__ __forceinline__ void atomicMaxF(float* addr, float val) {
  unsigned int* ua = (unsigned int*)addr;
  unsigned int old = *ua;
  while (__uint_as_float(old) < val) {
    unsigned int prev = atomicCAS(ua, old, __float_as_uint(val));
    if (prev == old) break;
    old = prev;
  }
}

// CDNA5 async global->LDS copy (tracked by ASYNCcnt).
// vdst VGPR carries the LDS byte address (flat-LDS addr[31:0]).
__device__ __forceinline__ void async_b128(uint32_t lds_off, const void* gaddr) {
  asm volatile("global_load_async_to_lds_b128 %0, %1, off"
               :: "v"(lds_off), "v"(gaddr) : "memory");
}

template<int N>
__device__ __forceinline__ void wait_async() {
#if __has_builtin(__builtin_amdgcn_s_wait_asynccnt)
  __builtin_amdgcn_s_wait_asynccnt(N);
#else
  asm volatile("s_wait_asynccnt %0" :: "i"(N));
#endif
}

// ---------------------------------------------------------------------------
// Generic f16 WMMA GEMM:  C[M][N] = A[M][K] @ Bt[N][K]^T   (f32 accumulate)
//   AMODE 0: A row-major [m][k], pitch lda (flat m across batches); async fill
//   AMODE 1: A given transposed At[k][m] per batch (pitch lda over m,
//            base = batch*bstrideA); transposing LDS scatter fill
//   Bt is [n][k] row-major, per-batch base = batch*bstrideB; async fill
//   EPI 0: store transposed f16: outT[b][n][v]  (v = m % 1024), pitch 1024
//   EPI 1: same, value = (acc + bias[n]) * mask[m]
//   EPI 2: val = (acc + bias[n]) * mask[m]; column-max -> atomicMax(pooled)
// Block: 256 threads (8 waves). Tile BM=128, BN=64, BK=32, double-buffered.
// ---------------------------------------------------------------------------
template<int AMODE, int EPI>
__global__ __launch_bounds__(256) void gemm_f16_wmma(
    const _Float16* __restrict__ A, long bstrideA, int lda,
    const _Float16* __restrict__ Bt, long bstrideB, int ldb,
    int K,
    _Float16* __restrict__ outT, long bstrideO,
    float* __restrict__ pooled, int poolN,
    const float* __restrict__ bias,
    const float* __restrict__ mask)
{
  __shared__ _Float16 As[2][128 * 32];  // [m][k] tiles, 2 x 8 KB
  __shared__ _Float16 Bs[2][64 * 32];   // [n][k] tiles, 2 x 4 KB

  const int tid  = threadIdx.x;
  const int lane = tid & 31;
  const int wave = tid >> 5;
  const int half = lane >> 4;
  const int l16  = lane & 15;
  const int m0 = blockIdx.x * 128;
  const int n0 = blockIdx.y * 64;
  const int batch = m0 >> 10;
  const int wm = (wave >> 1) * 32;
  const int wn = (wave & 1) * 32;

  const _Float16* Bb = Bt + (long)batch * bstrideB;
  const _Float16* Ab = (AMODE == 1) ? (A + (long)batch * bstrideA) : A;
  const int mloc = m0 & 1023;

  // # of async ops this thread issues per tile fill (in-order completion)
  constexpr int NFILL = (AMODE == 0) ? 3 : 1;

  // ---- per-thread fill of one LDS buffer for K-chunk k0 ----
  auto fill = [&](int buf, int k0) {
    // B tile (64x32): 1 chunk / thread, async
    {
      int nr = tid >> 2, kc = tid & 3;
      async_b128((uint32_t)(uintptr_t)&Bs[buf][nr * 32 + kc * 8],
                 Bb + (long)(n0 + nr) * ldb + k0 + kc * 8);
    }
    if (AMODE == 0) {
      // A tile (128x32): 2 chunks / thread, async
      #pragma unroll
      for (int j = 0; j < 2; j++) {
        int c = tid + 256 * j;
        int r = c >> 2, cc = c & 3;
        async_b128((uint32_t)(uintptr_t)&As[buf][r * 32 + cc * 8],
                   Ab + (long)(m0 + r) * lda + k0 + cc * 8);
      }
    } else {
      // A from At[k][m]: read 8 contiguous m, scatter-transpose into LDS
      #pragma unroll
      for (int j = 0; j < 2; j++) {
        int c = tid + 256 * j;
        int kr = c >> 4, mc = c & 15;
        uint4 d = *(const uint4*)(Ab + (long)(k0 + kr) * lda + mloc + mc * 8);
        const _Float16* e = (const _Float16*)&d;
        #pragma unroll
        for (int t = 0; t < 8; t++) As[buf][(mc * 8 + t) * 32 + kr] = e[t];
      }
    }
  };

  v8f zero = {0.f, 0.f, 0.f, 0.f, 0.f, 0.f, 0.f, 0.f};
  v8f acc[2][2];
  acc[0][0] = zero; acc[0][1] = zero; acc[1][0] = zero; acc[1][1] = zero;

  const int S = K >> 5;
  fill(0, 0);
  for (int s = 0; s < S; s++) {
    const int cur = s & 1;
    if (s + 1 < S) {
      fill(1 - cur, (s + 1) * 32);
      wait_async<NFILL>();   // in-order: first NFILL (cur tile) have landed
    } else {
      wait_async<0>();
    }
    __syncthreads();         // cur tile visible to all waves (ds waits folded in)

    Frag af[2], bf[2];
    #pragma unroll
    for (int i = 0; i < 2; i++) {
      int row = wm + i * 16 + l16;
      af[i].q[0] = *(const uint4*)(&As[cur][row * 32 + half * 8]);
      af[i].q[1] = *(const uint4*)(&As[cur][row * 32 + half * 8 + 16]);
    }
    #pragma unroll
    for (int j = 0; j < 2; j++) {
      int col = wn + j * 16 + l16;
      bf[j].q[0] = *(const uint4*)(&Bs[cur][col * 32 + half * 8]);
      bf[j].q[1] = *(const uint4*)(&Bs[cur][col * 32 + half * 8 + 16]);
    }
    #pragma unroll
    for (int i = 0; i < 2; i++)
      #pragma unroll
      for (int j = 0; j < 2; j++)
        acc[i][j] = __builtin_amdgcn_wmma_f32_16x16x32_f16(
            false, af[i].v, false, bf[j].v, (short)0, acc[i][j], false, false);
    __syncthreads();         // all waves done reading cur before it is refilled
  }

  // ---- epilogue ----
  if (EPI == 2) {
    #pragma unroll
    for (int j = 0; j < 2; j++) {
      int ncol = n0 + wn + j * 16 + l16;
      float bs = bias[ncol];
      float mx = -3.402823466e38f;
      #pragma unroll
      for (int i = 0; i < 2; i++) {
        int mbase = m0 + wm + i * 16 + half * 8;
        #pragma unroll
        for (int g = 0; g < 8; g++) {
          float v = (acc[i][j][g] + bs) * mask[mbase + g];
          mx = fmaxf(mx, v);
        }
      }
      atomicMaxF(&pooled[batch * poolN + ncol], mx);
    }
  } else {
    #pragma unroll
    for (int i = 0; i < 2; i++) {
      int mbase = m0 + wm + i * 16 + half * 8;
      int vbase = mbase & 1023;
      float mk[8];
      if (EPI == 1) {
        #pragma unroll
        for (int g = 0; g < 8; g++) mk[g] = mask[mbase + g];
      }
      #pragma unroll
      for (int j = 0; j < 2; j++) {
        int ncol = n0 + wn + j * 16 + l16;
        float bs = (EPI == 1) ? bias[ncol] : 0.0f;
        Pack8 p;
        #pragma unroll
        for (int g = 0; g < 8; g++) {
          float v = acc[i][j][g] + bs;
          if (EPI == 1) v *= mk[g];
          p.h[g] = (_Float16)v;
        }
        *(uint4*)(outT + (long)batch * bstrideO + (long)ncol * 1024 + vbase) = p.q;
      }
    }
  }
}

// ---------------------------------------------------------------------------
__global__ void cvt_f32_to_f16(const float* __restrict__ s,
                               _Float16* __restrict__ d, int n4) {
  int i = blockIdx.x * blockDim.x + threadIdx.x;
  int stride = gridDim.x * blockDim.x;
  for (int idx = i; idx < n4; idx += stride) {
    float4 f = ((const float4*)s)[idx];
    v4h h;
    h.x = (_Float16)f.x; h.y = (_Float16)f.y;
    h.z = (_Float16)f.z; h.w = (_Float16)f.w;
    ((v4h*)d)[idx] = h;
  }
}

__global__ void cvt_transpose_f16(const float* __restrict__ W,
                                  _Float16* __restrict__ Wt, int K, int N) {
  int idx = blockIdx.x * 256 + threadIdx.x;
  if (idx < K * N) {
    int k = idx / N, n = idx - k * N;
    Wt[n * K + k] = (_Float16)W[idx];
  }
}

__global__ void init_neg_inf(float* p, int n) {
  int i = blockIdx.x * 256 + threadIdx.x;
  if (i < n) p[i] = -3.402823466e38f;
}

__global__ void fc_kernel(const float* __restrict__ pooled,
                          const float* __restrict__ Wfc,
                          const float* __restrict__ bfc,
                          float* __restrict__ out) {
  int t = threadIdx.x;
  if (t >= 160) return;
  int b = t / 10, o = t - b * 10;
  float s = bfc[o];
  for (int f = 0; f < 128; f++) s += pooled[b * 128 + f] * Wfc[f * 10 + o];
  out[b * 10 + o] = s;
}

// ---------------------------------------------------------------------------
extern "C" void kernel_launch(void* const* d_in, const int* in_sizes, int n_in,
                              void* d_out, int out_size, void* d_ws, size_t ws_size,
                              hipStream_t stream) {
  const float* x    = (const float*)d_in[0];
  const float* adj  = (const float*)d_in[1];
  const float* mask = (const float*)d_in[2];
  const float* W1   = (const float*)d_in[3];
  const float* b1   = (const float*)d_in[4];
  const float* W2   = (const float*)d_in[5];
  const float* b2   = (const float*)d_in[6];
  const float* Wfc  = (const float*)d_in[7];
  const float* bfc  = (const float*)d_in[8];
  float* out = (float*)d_out;

  char* ws = (char*)d_ws;
  size_t off = 0;
  auto alloc = [&](size_t bytes) -> void* {
    void* p = ws + off;
    off += (bytes + 255) & ~(size_t)255;
    return p;
  };
  _Float16* adjh = (_Float16*)alloc((size_t)16 * 1024 * 1024 * 2); // 33.5 MB
  _Float16* xh   = (_Float16*)alloc((size_t)16 * 1024 * 256 * 2);  // 8 MB
  _Float16* w1t  = (_Float16*)alloc((size_t)256 * 256 * 2);        // W1^T [256n][256k]
  _Float16* w2t  = (_Float16*)alloc((size_t)128 * 256 * 2);        // W2^T [128n][256k]
  _Float16* yT   = (_Float16*)alloc((size_t)16 * 256 * 1024 * 2);  // (x@W1)^T per batch
  _Float16* h1T  = (_Float16*)alloc((size_t)16 * 256 * 1024 * 2);  // h1^T per batch
  _Float16* wT   = (_Float16*)alloc((size_t)16 * 128 * 1024 * 2);  // (h1@W2)^T per batch
  float*    pooled = (float*)alloc((size_t)16 * 128 * 4);

  cvt_f32_to_f16<<<4096, 256, 0, stream>>>(adj, adjh, 16 * 1024 * 1024 / 4);
  cvt_f32_to_f16<<<2048, 256, 0, stream>>>(x, xh, 16 * 1024 * 256 / 4);
  cvt_transpose_f16<<<(256 * 256 + 255) / 256, 256, 0, stream>>>(W1, w1t, 256, 256);
  cvt_transpose_f16<<<(256 * 128 + 255) / 256, 256, 0, stream>>>(W2, w2t, 256, 128);
  init_neg_inf<<<8, 256, 0, stream>>>(pooled, 16 * 128);

  dim3 blk(256);
  // y = x @ W1                    (M=16384, N=256, K=256) -> yT
  gemm_f16_wmma<0, 0><<<dim3(128, 4), blk, 0, stream>>>(
      xh, 0, 256, w1t, 0, 256, 256, yT, 256 * 1024, nullptr, 0, nullptr, nullptr);
  // h1 = (adj @ y + b1) * mask    (M=16384 flat, N=256, K=1024) -> h1T
  gemm_f16_wmma<0, 1><<<dim3(128, 4), blk, 0, stream>>>(
      adjh, 0, 1024, yT, 256 * 1024, 1024, 1024, h1T, 256 * 1024, nullptr, 0, b1, mask);
  // w = h1 @ W2                   (A = h1T col-major, N=128, K=256) -> wT
  gemm_f16_wmma<1, 0><<<dim3(128, 2), blk, 0, stream>>>(
      h1T, 256 * 1024, 1024, w2t, 0, 256, 256, wT, 128 * 1024, nullptr, 0, nullptr, nullptr);
  // pooled = max_v (adj @ w + b2) * mask   (N=128, K=1024)
  gemm_f16_wmma<0, 2><<<dim3(128, 2), blk, 0, stream>>>(
      adjh, 0, 1024, wT, 128 * 1024, 1024, 1024, nullptr, 0, pooled, 128, b2, mask);

  fc_kernel<<<1, 192, 0, stream>>>(pooled, Wfc, bfc, out);
}
